// DRKMamba_27178553049143
// MI455X (gfx1250) — compile-verified, matrix operand order checked by reading, not run on hardware
//
#include <hip/hip_runtime.h>
#include <hip/hip_bf16.h>

// ---------------- vector types ----------------
typedef _Float16 h8  __attribute__((ext_vector_type(8)));
typedef _Float16 h16 __attribute__((ext_vector_type(16)));
typedef float    v8f __attribute__((ext_vector_type(8)));
typedef int      v4i __attribute__((ext_vector_type(4)));

// ---------------- CDNA5 async global->LDS probe ----------------
#if defined(__has_builtin)
#if __has_builtin(__builtin_amdgcn_global_load_async_to_lds_b128)
#define HAS_ASYNC_LDS 1
#endif
#endif
#ifndef HAS_ASYNC_LDS
#define HAS_ASYNC_LDS 0
#endif

// builtin expects: (int4 global* src, int4 lds* dst, imm offset, imm cpol)
typedef __attribute__((address_space(1))) v4i gv4i;   // global int4
typedef __attribute__((address_space(3))) v4i lv4i;   // LDS int4

__device__ __forceinline__ void async_cp16(const void* g, void* l) {
#if HAS_ASYNC_LDS
    __builtin_amdgcn_global_load_async_to_lds_b128((gv4i*)g, (lv4i*)l, 0, 0);
#else
    (void)g; (void)l;
#endif
}

__device__ __forceinline__ void wait_async_all() {
#if HAS_ASYNC_LDS
#if __has_builtin(__builtin_amdgcn_s_wait_asynccnt)
    __builtin_amdgcn_s_wait_asynccnt(0);
#else
    asm volatile("s_wait_asynccnt 0x0" ::: "memory");
#endif
#endif
}

// ---------------- device math helpers ----------------
__device__ __forceinline__ float dev_silu(float x) {
    return x / (1.f + __expf(-x));
}
__device__ __forceinline__ float dev_gelu(float x) {
    float x3 = x * x * x;
    return 0.5f * x * (1.f + tanhf(0.7978845608028654f * (x + 0.044715f * x3)));
}
__device__ __forceinline__ float dev_softplus(float x) {
    return (x > 20.f) ? x : log1pf(__expf(x));
}

// Load one WMMA 16x16x32 f16 fragment slice from LDS.
// Per-lane layout (A and B^T identical): halves [k0..k0+7] then [k0+16..k0+23],
// k0 = (lane>=16)*8.  `p` points at (row*32 + k0) within the tile.
__device__ __forceinline__ h16 frag_ld(const _Float16* p) {
    union { h16 v; h8 p8[2]; } u;
    u.p8[0] = *(const h8*)(p);
    u.p8[1] = *(const h8*)(p + 16);
    return u.v;
}

#define WMMA(A_, B_, C_) \
    __builtin_amdgcn_wmma_f32_16x16x32_f16(false, (A_), false, (B_), (short)0, (C_), false, false)

// =====================================================================
// Generic WMMA GEMM:  Co[M x N] = A[M x Kpad](f16) * Bt[Npad x Kpad]^T(f16)
//   + optional bias, + optional residual, + optional softplus.
// Block: 128 threads (4 waves). Block tile 128x64; each wave 32 rows x 64 cols
// (2 M-frags x 4 N-frags = 8 WMMA per K-step). Double-buffered LDS,
// async global->LDS copies when the toolchain exposes them.
// M must be a multiple of 128; Kpad multiple of 32; Npad = ceil64(N).
// =====================================================================
__global__ __launch_bounds__(128) void k_gemm(
    const _Float16* __restrict__ A, int lda,
    const _Float16* __restrict__ Bt, int ldb,
    float* __restrict__ Co, int ldc,
    const float* __restrict__ bias,
    const float* __restrict__ resid, int ldr,
    int M, int N, int Kpad, int act)
{
    (void)M;
    __shared__ _Float16 As[2][128 * 32];
    __shared__ _Float16 Bs[2][64 * 32];
    const int tid  = threadIdx.x;
    const int lane = tid & 31;
    const int wave = tid >> 5;
    const int bm = blockIdx.x * 128;
    const int bn = blockIdx.y * 64;
    const int brow = tid >> 1;              // 0..63 B row
    const int bcol = (tid & 1) * 16;        // 0 / 16 halves

    v8f acc0 = {}, acc1 = {}, acc2 = {}, acc3 = {};
    v8f acc4 = {}, acc5 = {}, acc6 = {}, acc7 = {};

    auto load_tile = [&](int kk, int buf) {
        const _Float16* ga = A + (size_t)(bm + tid) * lda + kk;     // row = tid, 32 halves
        _Float16* da = &As[buf][tid * 32];
        const _Float16* gb = Bt + (size_t)(bn + brow) * ldb + kk + bcol;
        _Float16* db = &Bs[buf][brow * 32 + bcol];
#if HAS_ASYNC_LDS
        async_cp16(ga,      da);
        async_cp16(ga + 8,  da + 8);
        async_cp16(ga + 16, da + 16);
        async_cp16(ga + 24, da + 24);
        async_cp16(gb,      db);
        async_cp16(gb + 8,  db + 8);
#else
        *(h8*)(da)      = *(const h8*)(ga);
        *(h8*)(da + 8)  = *(const h8*)(ga + 8);
        *(h8*)(da + 16) = *(const h8*)(ga + 16);
        *(h8*)(da + 24) = *(const h8*)(ga + 24);
        *(h8*)(db)      = *(const h8*)(gb);
        *(h8*)(db + 8)  = *(const h8*)(gb + 8);
        __builtin_prefetch(ga + 32, 0, 3);
        __builtin_prefetch(gb + 32, 0, 3);
#endif
    };

    const int nk = Kpad / 32;
    load_tile(0, 0);

    for (int s = 0; s < nk; ++s) {
        wait_async_all();          // own async writes visible before barrier
        __syncthreads();           // all waves' tile[s] writes visible
        if (s + 1 < nk) load_tile((s + 1) * 32, (s + 1) & 1);

        const int k0 = (lane >> 4) * 8;
        const int rl = lane & 15;
        const _Float16* ab = &As[s & 1][0];
        const _Float16* bb = &Bs[s & 1][0];
        h16 a0 = frag_ld(ab + (32 * wave + rl) * 32 + k0);
        h16 a1 = frag_ld(ab + (32 * wave + 16 + rl) * 32 + k0);
        h16 b0 = frag_ld(bb + (rl)      * 32 + k0);
        h16 b1 = frag_ld(bb + (16 + rl) * 32 + k0);
        h16 b2 = frag_ld(bb + (32 + rl) * 32 + k0);
        h16 b3 = frag_ld(bb + (48 + rl) * 32 + k0);
        acc0 = WMMA(a0, b0, acc0);  acc1 = WMMA(a0, b1, acc1);
        acc2 = WMMA(a0, b2, acc2);  acc3 = WMMA(a0, b3, acc3);
        acc4 = WMMA(a1, b0, acc4);  acc5 = WMMA(a1, b1, acc5);
        acc6 = WMMA(a1, b2, acc6);  acc7 = WMMA(a1, b3, acc7);
        __syncthreads();           // finish reads of tile[s] before it is reloaded
    }

    const int cl = lane & 15;

#define EPI(ACC, MT, J) do {                                               \
        int col = bn + 16 * (J) + cl;                                      \
        if (col < N) {                                                     \
            int rb = bm + 32 * wave + 16 * (MT) + ((lane >> 4) << 3);      \
            _Pragma("unroll")                                              \
            for (int r = 0; r < 8; ++r) {                                  \
                int row = rb + r;                                          \
                float v = (ACC)[r];                                        \
                if (bias)  v += bias[col];                                 \
                if (resid) v += resid[(size_t)row * ldr + col];            \
                if (act == 1) v = dev_softplus(v);                         \
                Co[(size_t)row * ldc + col] = v;                           \
            }                                                              \
        }                                                                  \
    } while (0)

    EPI(acc0, 0, 0); EPI(acc1, 0, 1); EPI(acc2, 0, 2); EPI(acc3, 0, 3);
    EPI(acc4, 1, 0); EPI(acc5, 1, 1); EPI(acc6, 1, 2); EPI(acc7, 1, 3);
#undef EPI
}

// =====================================================================
// Conv(K=5, SAME) as WMMA GEMM with fused bias + maxpool2.
// A:  [Lin x Kpad] f16 (LN+GELU staged).  Bt: [5 x Npad x Kpad] f16.
// Co: [Lin/2 x ldc] f32 pooled output.
// Same 128x64 double-buffered tiling; sync loads (boundary taps need
// zero fill).  Pool fusion: rows (2m,2m+1) live in adjacent accum VGPRs.
// =====================================================================
__global__ __launch_bounds__(128) void k_conv_gemm(
    const _Float16* __restrict__ A,
    const _Float16* __restrict__ Bt,
    float* __restrict__ Co, int ldc,
    const float* __restrict__ bias,
    int Lin, int N, int Kpad, int Npad)
{
    __shared__ _Float16 As[2][128 * 32];
    __shared__ _Float16 Bs[2][64 * 32];
    const int tid  = threadIdx.x;
    const int lane = tid & 31;
    const int wave = tid >> 5;
    const int bm = blockIdx.x * 128;
    const int bn = blockIdx.y * 64;
    const int brow = tid >> 1;
    const int bcol = (tid & 1) * 16;
    const int nk = Kpad / 32;

    v8f acc0 = {}, acc1 = {}, acc2 = {}, acc3 = {};
    v8f acc4 = {}, acc5 = {}, acc6 = {}, acc7 = {};

    auto load_tile = [&](int s, int buf) {
        const int t  = s / nk;
        const int kk = (s - t * nk) * 32;
        const int ar = bm + tid + t - 2;           // SAME padding, K=5
        _Float16* da = &As[buf][tid * 32];
        if (ar >= 0 && ar < Lin) {
            const _Float16* ga = A + (size_t)ar * Kpad + kk;
            *(h8*)(da)      = *(const h8*)(ga);
            *(h8*)(da + 8)  = *(const h8*)(ga + 8);
            *(h8*)(da + 16) = *(const h8*)(ga + 16);
            *(h8*)(da + 24) = *(const h8*)(ga + 24);
        } else {
            h8 z = {};
            *(h8*)(da) = z; *(h8*)(da + 8) = z; *(h8*)(da + 16) = z; *(h8*)(da + 24) = z;
        }
        const _Float16* gb = Bt + ((size_t)t * Npad + bn + brow) * Kpad + kk + bcol;
        _Float16* db = &Bs[buf][brow * 32 + bcol];
        *(h8*)(db)     = *(const h8*)(gb);
        *(h8*)(db + 8) = *(const h8*)(gb + 8);
    };

    const int ns = 5 * nk;
    load_tile(0, 0);

    for (int s = 0; s < ns; ++s) {
        __syncthreads();
        if (s + 1 < ns) load_tile(s + 1, (s + 1) & 1);
        const int k0 = (lane >> 4) * 8;
        const int rl = lane & 15;
        const _Float16* ab = &As[s & 1][0];
        const _Float16* bb = &Bs[s & 1][0];
        h16 a0 = frag_ld(ab + (32 * wave + rl) * 32 + k0);
        h16 a1 = frag_ld(ab + (32 * wave + 16 + rl) * 32 + k0);
        h16 b0 = frag_ld(bb + (rl)      * 32 + k0);
        h16 b1 = frag_ld(bb + (16 + rl) * 32 + k0);
        h16 b2 = frag_ld(bb + (32 + rl) * 32 + k0);
        h16 b3 = frag_ld(bb + (48 + rl) * 32 + k0);
        acc0 = WMMA(a0, b0, acc0);  acc1 = WMMA(a0, b1, acc1);
        acc2 = WMMA(a0, b2, acc2);  acc3 = WMMA(a0, b3, acc3);
        acc4 = WMMA(a1, b0, acc4);  acc5 = WMMA(a1, b1, acc5);
        acc6 = WMMA(a1, b2, acc6);  acc7 = WMMA(a1, b3, acc7);
        __syncthreads();
    }

    const int cl = lane & 15;

#define CEPI(ACC, MT, J) do {                                              \
        int col = bn + 16 * (J) + cl;                                      \
        if (col < N) {                                                     \
            int rb = bm + 32 * wave + 16 * (MT) + ((lane >> 4) << 3);      \
            _Pragma("unroll")                                              \
            for (int r2 = 0; r2 < 4; ++r2) {                               \
                int row = rb + 2 * r2;                                     \
                float v = fmaxf((ACC)[2 * r2], (ACC)[2 * r2 + 1]) + bias[col]; \
                Co[(size_t)(row >> 1) * ldc + col] = v;                    \
            }                                                              \
        }                                                                  \
    } while (0)

    CEPI(acc0, 0, 0); CEPI(acc1, 0, 1); CEPI(acc2, 0, 2); CEPI(acc3, 0, 3);
    CEPI(acc4, 1, 0); CEPI(acc5, 1, 1); CEPI(acc6, 1, 2); CEPI(acc7, 1, 3);
#undef CEPI
}

// =====================================================================
// DNA conv: one-hot (32768 x 4) -> conv(K=9, SAME, 4->320) + bias -> maxpool2
// =====================================================================
__global__ void k_dna(const float* __restrict__ X, const float* __restrict__ W,
                      const float* __restrict__ Bv, float* __restrict__ Out)
{
    int idx = blockIdx.x * blockDim.x + threadIdx.x;
    if (idx >= 16384 * 320) return;
    int c = idx % 320, p = idx / 320;
    float a0 = Bv[c], a1 = Bv[c];
#pragma unroll
    for (int t = 0; t < 9; ++t) {
        int p0 = 2 * p + t - 4;
        int p1 = p0 + 1;
#pragma unroll
        for (int i = 0; i < 4; ++i) {
            float w = W[(t * 4 + i) * 320 + c];
            if ((unsigned)p0 < 32768u) a0 += X[p0 * 4 + i] * w;
            if ((unsigned)p1 < 32768u) a1 += X[p1 * 4 + i] * w;
        }
    }
    Out[idx] = fmaxf(a0, a1);
}

// =====================================================================
// LayerNorm (+ optional GELU) over channels, write f16 padded staging.
// =====================================================================
__global__ __launch_bounds__(256) void k_ln_act_h(
    const float* __restrict__ X, const float* __restrict__ g,
    const float* __restrict__ b, _Float16* __restrict__ Y,
    int C, int Cpad, int act)
{
    const int row = blockIdx.x;
    const int tid = threadIdx.x;
    __shared__ float red[256];

    float s = 0.f;
    for (int c = tid; c < C; c += 256) s += X[(size_t)row * C + c];
    red[tid] = s; __syncthreads();
    for (int k = 128; k; k >>= 1) { if (tid < k) red[tid] += red[tid + k]; __syncthreads(); }
    const float mean = red[0] / (float)C;
    __syncthreads();

    float v = 0.f;
    for (int c = tid; c < C; c += 256) {
        float d = X[(size_t)row * C + c] - mean;
        v += d * d;
    }
    red[tid] = v; __syncthreads();
    for (int k = 128; k; k >>= 1) { if (tid < k) red[tid] += red[tid + k]; __syncthreads(); }
    const float rstd = rsqrtf(red[0] / (float)C + 1e-6f);
    __syncthreads();

    for (int c = tid; c < Cpad; c += 256) {
        float val = 0.f;
        if (c < C) {
            val = (X[(size_t)row * C + c] - mean) * rstd * g[c] + b[c];
            if (act) val = dev_gelu(val);
        }
        Y[(size_t)row * Cpad + c] = (_Float16)val;
    }
}

// =====================================================================
// Weight prep: f32 W[taps x K x N] -> f16 Wt[taps x Npad x Kpad] (B^T, zero pad)
// =====================================================================
__global__ void k_prep_wt(const float* __restrict__ W, _Float16* __restrict__ Wt,
                          int taps, int K, int N, int Kpad, int Npad)
{
    size_t idx = (size_t)blockIdx.x * blockDim.x + threadIdx.x;
    size_t tot = (size_t)taps * Npad * Kpad;
    if (idx >= tot) return;
    int k = (int)(idx % Kpad);
    size_t r = idx / Kpad;
    int n = (int)(r % Npad);
    int t = (int)(r / Npad);
    float v = (k < K && n < N) ? W[((size_t)t * K + k) * N + n] : 0.f;
    Wt[idx] = (_Float16)v;
}

// =====================================================================
// Mamba depthwise causal conv (K=4) + SiLU; optional sequence reversal.
// =====================================================================
__global__ void k_dwconv_silu(const float* __restrict__ XZ, const float* __restrict__ CW,
                              const float* __restrict__ CB, float* __restrict__ XS,
                              _Float16* __restrict__ XSH, int L, int E, int rev)
{
    int idx = blockIdx.x * blockDim.x + threadIdx.x;
    if (idx >= L * E) return;
    int e = idx % E, l = idx / E;
    float acc = CB[e];
#pragma unroll
    for (int t = 0; t < 4; ++t) {
        int lu = l - 3 + t;
        if (lu >= 0) {
            int ls = rev ? (L - 1 - lu) : lu;
            acc += XZ[(size_t)ls * 3072 + e] * CW[t * E + e];
        }
    }
    float s = dev_silu(acc);
    XS[idx]  = s;
    XSH[idx] = (_Float16)s;
}

// proj f32 (L x 64) -> f16 A-staging with cols >= 48 zeroed (dt GEMM input)
__global__ void k_projprep(const float* __restrict__ P, _Float16* __restrict__ D)
{
    int idx = blockIdx.x * blockDim.x + threadIdx.x;
    if (idx >= 1024 * 64) return;
    int c = idx % 64;
    D[idx] = (c < 48) ? (_Float16)P[idx] : (_Float16)0.f;
}

// =====================================================================
// Selective scan. One thread per channel e, 8 states in registers,
// B/C (proj cols 48..63) staged in LDS per 64-step chunk.
// =====================================================================
__global__ __launch_bounds__(128) void k_scan(
    const float* __restrict__ XS, const float* __restrict__ DT,
    const float* __restrict__ PJ, const float* __restrict__ Alog,
    const float* __restrict__ Dv, float* __restrict__ Y,
    int L, int E, int rev, int accum)
{
    const int e = blockIdx.x * 128 + threadIdx.x;
    float a[8], h[8];
#pragma unroll
    for (int n = 0; n < 8; ++n) { a[n] = -__expf(Alog[(size_t)e * 8 + n]); h[n] = 0.f; }
    const float dcoef = Dv[e];
    __shared__ float bc[64 * 16];

    for (int l0 = 0; l0 < L; l0 += 64) {
        for (int i = threadIdx.x; i < 64 * 16; i += 128)
            bc[i] = PJ[(size_t)(l0 + (i >> 4)) * 64 + 48 + (i & 15)];
        __syncthreads();
        for (int l = l0; l < l0 + 64; ++l) {
            float dtv = DT[(size_t)l * E + e];
            float xv  = XS[(size_t)l * E + e];
            const float* Bp = &bc[(l - l0) * 16];
            float yv = 0.f;
#pragma unroll
            for (int n = 0; n < 8; ++n) {
                h[n] = __expf(dtv * a[n]) * h[n] + dtv * Bp[n] * xv;
                yv += h[n] * Bp[8 + n];
            }
            yv += xv * dcoef;
            int lw = rev ? (L - 1 - l) : l;
            size_t oi = (size_t)lw * E + e;
            if (accum) Y[oi] += yv; else Y[oi] = yv;
        }
        __syncthreads();
    }
}

// y_total * silu(z) -> f16 staging for out_proj
__global__ void k_gate(const float* __restrict__ Y, const float* __restrict__ XZ,
                       _Float16* __restrict__ YG, int L, int E)
{
    int idx = blockIdx.x * blockDim.x + threadIdx.x;
    if (idx >= L * E) return;
    int e = idx % E, l = idx / E;
    float z = XZ[(size_t)l * 3072 + 1536 + e];
    YG[idx] = (_Float16)(Y[idx] * dev_silu(z));
}

// head staging: crop rows 256..767, GELU, cast f16
__global__ void k_head_stage(const float* __restrict__ Xin, _Float16* __restrict__ O)
{
    int idx = blockIdx.x * blockDim.x + threadIdx.x;
    if (idx >= 512 * 768) return;
    int c = idx % 768, r = idx / 768;
    O[idx] = (_Float16)dev_gelu(Xin[(size_t)(256 + r) * 768 + c]);
}

// =====================================================================
// Host launcher
// =====================================================================
extern "C" void kernel_launch(void* const* d_in, const int* in_sizes, int n_in,
                              void* d_out, int out_size, void* d_ws, size_t ws_size,
                              hipStream_t stream) {
    (void)in_sizes; (void)n_in; (void)out_size; (void)ws_size;
    auto F = [&](int i) -> const float* { return (const float*)d_in[i]; };

    // ---- input unpacking (setup_inputs dict order, depth-first) ----
    int i = 0;
    const float* X = F(i++);                       // (1,32768,4)
    const float* dna_w = F(i++); const float* dna_b = F(i++);
    const float *tg[4], *tb[4], *tw[4], *tbias[4];
    for (int k = 0; k < 4; ++k) { tg[k] = F(i++); tb[k] = F(i++); tw[k] = F(i++); tbias[k] = F(i++); }
    struct Dir { const float *cw, *cb, *xp, *dtw, *dtb, *alog, *dv; };
    struct Lay { const float *lg, *lb, *inp, *outp; Dir d[2]; } lay[2];
    for (int l = 0; l < 2; ++l) {
        lay[l].lg = F(i++); lay[l].lb = F(i++); lay[l].inp = F(i++); lay[l].outp = F(i++);
        for (int dd = 0; dd < 2; ++dd) {
            lay[l].d[dd].cw = F(i++); lay[l].d[dd].cb = F(i++); lay[l].d[dd].xp = F(i++);
            lay[l].d[dd].dtw = F(i++); lay[l].d[dd].dtb = F(i++);
            lay[l].d[dd].alog = F(i++); lay[l].d[dd].dv = F(i++);
        }
    }
    const float* hw = F(i++); const float* hb = F(i++);

    // ---- workspace bump allocator ----
    char* wsb = (char*)d_ws; size_t off = 0;
    auto alloc = [&](size_t bytes) -> void* {
        void* p = wsb + off; off += (bytes + 255) & ~(size_t)255; return p;
    };

    float* actA = (float*)alloc((size_t)16384 * 320 * 4);
    float* actB = (float*)alloc((size_t)16384 * 320 * 4);
    _Float16* xh   = (_Float16*)alloc((size_t)16384 * 320 * 2);   // max A-staging
    float* xz      = (float*)alloc((size_t)1024 * 3072 * 4);
    float* xs      = (float*)alloc((size_t)1024 * 1536 * 4);
    _Float16* xs_h = (_Float16*)alloc((size_t)1024 * 1536 * 2);
    float* proj    = (float*)alloc((size_t)1024 * 64 * 4);
    _Float16* dtA  = (_Float16*)alloc((size_t)1024 * 64 * 2);
    float* dtf     = (float*)alloc((size_t)1024 * 1536 * 4);
    float* yacc    = (float*)alloc((size_t)1024 * 1536 * 4);
    _Float16* yg   = (_Float16*)alloc((size_t)1024 * 1536 * 2);

    // prepped f16 B^T weights
    const int Cin[4]   = {320, 384, 484, 610}, Cinp[4]  = {320, 384, 512, 640};
    const int Coutv[4] = {384, 484, 610, 768}, Coutp[4] = {384, 512, 640, 768};
    _Float16* wtT[4];
    for (int k = 0; k < 4; ++k) wtT[k] = (_Float16*)alloc((size_t)5 * Coutp[k] * Cinp[k] * 2);
    _Float16 *wt_in[2], *wt_out[2], *wt_xp[2][2], *wt_dt[2][2];
    for (int l = 0; l < 2; ++l) {
        wt_in[l]  = (_Float16*)alloc((size_t)3072 * 768 * 2);
        wt_out[l] = (_Float16*)alloc((size_t)768 * 1536 * 2);
        for (int dd = 0; dd < 2; ++dd) {
            wt_xp[l][dd] = (_Float16*)alloc((size_t)64 * 1536 * 2);
            wt_dt[l][dd] = (_Float16*)alloc((size_t)1536 * 64 * 2);
        }
    }
    _Float16* wt_head = (_Float16*)alloc((size_t)512 * 768 * 2);

    auto prep = [&](const float* W, _Float16* Wt, int taps, int K, int N, int Kpad, int Npad) {
        size_t tot = (size_t)taps * Npad * Kpad;
        k_prep_wt<<<(unsigned)((tot + 255) / 256), 256, 0, stream>>>(W, Wt, taps, K, N, Kpad, Npad);
    };

    // ---- weight preparation ----
    for (int k = 0; k < 4; ++k)
        prep(tw[k], wtT[k], 5, Cin[k], Coutv[k], Cinp[k], Coutp[k]);
    for (int l = 0; l < 2; ++l) {
        prep(lay[l].inp,  wt_in[l],  1, 768, 3072, 768, 3072);
        prep(lay[l].outp, wt_out[l], 1, 1536, 768, 1536, 768);
        for (int dd = 0; dd < 2; ++dd) {
            prep(lay[l].d[dd].xp,  wt_xp[l][dd], 1, 1536, 64, 1536, 64);
            prep(lay[l].d[dd].dtw, wt_dt[l][dd], 1, 48, 1536, 64, 1536);
        }
    }
    prep(hw, wt_head, 1, 768, 512, 768, 512);

    // ---- trunk ----
    k_dna<<<(16384 * 320 + 255) / 256, 256, 0, stream>>>(X, dna_w, dna_b, actA);

    int Lcur = 16384;
    float* cur = actA; float* nxt = actB;
    for (int k = 0; k < 4; ++k) {
        k_ln_act_h<<<Lcur, 256, 0, stream>>>(cur, tg[k], tb[k], xh, Cin[k], Cinp[k], 1);
        dim3 gc(Lcur / 128, (Coutv[k] + 63) / 64);
        k_conv_gemm<<<gc, 128, 0, stream>>>(xh, wtT[k], nxt, Coutv[k], tbias[k],
                                            Lcur, Coutv[k], Cinp[k], Coutp[k]);
        Lcur /= 2;
        float* t = cur; cur = nxt; nxt = t;
    }
    // cur = trunk output: 1024 x 768

    // ---- mamba blocks ----
    float* xbuf = cur;
    for (int l = 0; l < 2; ++l) {
        k_ln_act_h<<<1024, 256, 0, stream>>>(xbuf, lay[l].lg, lay[l].lb, xh, 768, 768, 0);
        { dim3 g(1024 / 128, 3072 / 64);
          k_gemm<<<g, 128, 0, stream>>>(xh, 768, wt_in[l], 768, xz, 3072,
                                        nullptr, nullptr, 0, 1024, 3072, 768, 0); }
        for (int dd = 0; dd < 2; ++dd) {
            k_dwconv_silu<<<(1024 * 1536 + 255) / 256, 256, 0, stream>>>(
                xz, lay[l].d[dd].cw, lay[l].d[dd].cb, xs, xs_h, 1024, 1536, dd);
            { dim3 g(1024 / 128, 1);
              k_gemm<<<g, 128, 0, stream>>>(xs_h, 1536, wt_xp[l][dd], 1536, proj, 64,
                                            nullptr, nullptr, 0, 1024, 64, 1536, 0); }
            k_projprep<<<(1024 * 64 + 255) / 256, 256, 0, stream>>>(proj, dtA);
            { dim3 g(1024 / 128, 1536 / 64);
              k_gemm<<<g, 128, 0, stream>>>(dtA, 64, wt_dt[l][dd], 64, dtf, 1536,
                                            lay[l].d[dd].dtb, nullptr, 0, 1024, 1536, 64, 1); }
            k_scan<<<1536 / 128, 128, 0, stream>>>(xs, dtf, proj, lay[l].d[dd].alog,
                                                   lay[l].d[dd].dv, yacc, 1024, 1536, dd, dd);
        }
        k_gate<<<(1024 * 1536 + 255) / 256, 256, 0, stream>>>(yacc, xz, yg, 1024, 1536);
        { dim3 g(1024 / 128, 768 / 64);
          k_gemm<<<g, 128, 0, stream>>>(yg, 1536, wt_out[l], 1536, xbuf, 768,
                                        nullptr, xbuf, 768, 1024, 768, 1536, 0); }
    }

    // ---- head ----
    k_head_stage<<<(512 * 768 + 255) / 256, 256, 0, stream>>>(xbuf, xh);
    { dim3 g(512 / 128, 512 / 64);
      k_gemm<<<g, 128, 0, stream>>>(xh, 768, wt_head, 768, (float*)d_out, 512,
                                    hb, nullptr, 0, 512, 512, 768, 1); }
}
